// CENBase_13795434955242
// MI455X (gfx1250) — matrix operand order repaired and checked by reading, not compiled, not run on hardware
//
#include <hip/hip_runtime.h>
#include <math.h>

// Problem constants (from the reference)
#define N_ENT 20000
#define R2C   460
#define DIM   200
#define CCH   50
#define KW    3
#define LLY   10
#define NE    40000
#define NQ    2048

typedef __attribute__((ext_vector_type(16))) __bf16 v16bf;
typedef __attribute__((ext_vector_type(8)))  float  v8f;
typedef __attribute__((ext_vector_type(4)))  unsigned int v4u;
typedef __attribute__((ext_vector_type(8)))  int v8i_;
typedef __attribute__((ext_vector_type(4)))  int v4i_;

union FragU { v16bf v; unsigned int u[8]; };

#if defined(__has_builtin)
#  if __has_builtin(__builtin_amdgcn_tensor_load_to_lds)
#    define HAVE_TDM 1
#  else
#    define HAVE_TDM 0
#  endif
#else
#  define HAVE_TDM 0
#endif

static __device__ __forceinline__ unsigned short f32_to_bf16(float f) {
  unsigned int u = __float_as_uint(f);
  u += 0x7fffu + ((u >> 16) & 1u);   // round-to-nearest-even
  return (unsigned short)(u >> 16);
}
static __device__ __forceinline__ unsigned int pack_bf16(float lo, float hi) {
  return (unsigned int)f32_to_bf16(lo) | ((unsigned int)f32_to_bf16(hi) << 16);
}

#if HAVE_TDM
// LDS byte offset of a __shared__ object: flat LDS aperture keeps the offset
// in the low 32 address bits (ISA 10.2).
static __device__ __forceinline__ unsigned lds_off(const void* p) {
  return (unsigned)(unsigned long long)p;
}
// Issue one 2D TDM tile load: tile (tile1 rows x tile0 cols, fp32) from a
// row-major tensor slice starting at gptr (row stride = stride0 elements).
// tensor_dim0/1 = remaining extent from tile origin -> TDM zero-fills OOB.
static __device__ __forceinline__ void tdm_load_tile(const float* gptr, unsigned ldsaddr,
                                                     unsigned td0, unsigned td1,
                                                     unsigned tile0, unsigned tile1,
                                                     unsigned stride0) {
  unsigned long long ga = (unsigned long long)gptr;
  v4u g0;
  g0.x = 1u;                                         // count=1 (valid user D#)
  g0.y = ldsaddr;                                    // lds_addr [63:32]
  g0.z = (unsigned)(ga & 0xffffffffu);               // global_addr[31:0]
  g0.w = (unsigned)((ga >> 32) & 0x01ffffffu) | 0x80000000u;  // addr[56:32] | type=2
  v8i_ g1;
  g1[0] = (int)(2u << 16);                           // data_size = 4 bytes
  g1[1] = (int)((td0 & 0xffffu) << 16);              // tensor_dim0 lo16 @bit48
  g1[2] = (int)((td0 >> 16) | ((td1 & 0xffffu) << 16));
  g1[3] = (int)((td1 >> 16) | (tile0 << 16));        // tile_dim0 @bit112
  g1[4] = (int)(tile1 & 0xffffu);                    // tile_dim1; tile_dim2=0
  g1[5] = (int)stride0;                              // tensor_dim0_stride lo32
  g1[6] = 0;
  g1[7] = 0;
  v4i_ gz = {0, 0, 0, 0};
#if __clang_major__ >= 23
  v8i_ gz8 = {0, 0, 0, 0, 0, 0, 0, 0};
  __builtin_amdgcn_tensor_load_to_lds(g0, g1, gz, gz, gz8, 0);
#else
  __builtin_amdgcn_tensor_load_to_lds(g0, g1, gz, gz, 0);
#endif
}
#endif  // HAVE_TDM

// ---------------------------------------------------------------------------
// bf16 WMMA GEMM: C[M,N] = (acc ? C : 0) + A[M,K] * op(B)
//   A row-major fp32 (lda); B row-major fp32 (ldb); transB -> B is [N,K].
//   128 threads = 4 waves. Block tile 64x64, wave tile 32x32 (4 WMMA / K-step).
//   TDM path: raw fp32 tiles DMA'd to LDS (double buffered, TENSORcnt),
//   then converted LDS->LDS into bf16 fragment layouts As[m][k] / BsT[n][k].
// ---------------------------------------------------------------------------
__global__ void gemm_bf16w(const float* __restrict__ A, const float* __restrict__ B,
                           float* __restrict__ Cmat, int M, int Nn, int Kd,
                           int lda, int ldb, int ldc, int transB, int accumulate) {
  __shared__ unsigned short As[64][34];
  __shared__ unsigned short BsT[64][34];
  const int tid  = threadIdx.x;
  const int lane = tid & 31;
  const int wv   = tid >> 5;
  const int m0 = blockIdx.y * 64;
  const int n0 = blockIdx.x * 64;
  const int wm = (wv >> 1) * 32;
  const int wn = (wv & 1) * 32;

  const v8f z8 = {0.f, 0.f, 0.f, 0.f, 0.f, 0.f, 0.f, 0.f};
  v8f acc[2][2] = {{z8, z8}, {z8, z8}};

#if HAVE_TDM
  __shared__ __align__(16) float Araw[2][64 * 32];
  __shared__ __align__(16) float Braw[2][64 * 32];
  // Preload K-step 0 into buffer 0 (wave 0 drives the TDM; EXEC is ignored).
  if (wv == 0) {
    tdm_load_tile(A + (size_t)m0 * lda, lds_off(&Araw[0][0]),
                  (unsigned)Kd, (unsigned)(M - m0), 32u, 64u, (unsigned)lda);
    if (transB)
      tdm_load_tile(B + (size_t)n0 * ldb, lds_off(&Braw[0][0]),
                    (unsigned)Kd, (unsigned)(Nn - n0), 32u, 64u, (unsigned)ldb);
    else
      tdm_load_tile(B + n0, lds_off(&Braw[0][0]),
                    (unsigned)(Nn - n0), (unsigned)Kd, 64u, 32u, (unsigned)ldb);
    __builtin_amdgcn_s_wait_tensorcnt(0);
  }
  __syncthreads();
  int buf = 0;
#endif

  for (int kt = 0; kt < Kd; kt += 32) {
#if HAVE_TDM
    // Issue DMA for the next K-step into the other buffer (overlaps compute).
    if ((kt + 32 < Kd) && wv == 0) {
      int kn = kt + 32;
      tdm_load_tile(A + (size_t)m0 * lda + kn, lds_off(&Araw[buf ^ 1][0]),
                    (unsigned)(Kd - kn), (unsigned)(M - m0), 32u, 64u, (unsigned)lda);
      if (transB)
        tdm_load_tile(B + (size_t)n0 * ldb + kn, lds_off(&Braw[buf ^ 1][0]),
                      (unsigned)(Kd - kn), (unsigned)(Nn - n0), 32u, 64u, (unsigned)ldb);
      else
        tdm_load_tile(B + (size_t)kn * ldb + n0, lds_off(&Braw[buf ^ 1][0]),
                      (unsigned)(Nn - n0), (unsigned)(Kd - kn), 64u, 32u, (unsigned)ldb);
    }
    // Convert current raw fp32 tiles to bf16 fragment layouts (no guards:
    // TDM already zero-filled out-of-range elements).
#pragma unroll
    for (int s = 0; s < 4; ++s) {
      int lin = s * 128 + tid;
      int r   = lin >> 3;                 // m-local 0..63
      int c4  = (lin & 7) << 2;           // k-local
      const float4 vv = *(const float4*)&Araw[buf][r * 32 + c4];
      *(unsigned int*)&As[r][c4]     = pack_bf16(vv.x, vv.y);
      *(unsigned int*)&As[r][c4 + 2] = pack_bf16(vv.z, vv.w);
    }
    if (transB) {
#pragma unroll
      for (int s = 0; s < 4; ++s) {
        int lin = s * 128 + tid;
        int r   = lin >> 3;               // n-local 0..63
        int c4  = (lin & 7) << 2;         // k-local
        const float4 vv = *(const float4*)&Braw[buf][r * 32 + c4];
        *(unsigned int*)&BsT[r][c4]     = pack_bf16(vv.x, vv.y);
        *(unsigned int*)&BsT[r][c4 + 2] = pack_bf16(vv.z, vv.w);
      }
    } else {
#pragma unroll
      for (int s = 0; s < 4; ++s) {
        int lin = s * 128 + tid;
        int r   = lin >> 4;               // k-local 0..31
        int c4  = (lin & 15) << 2;        // n-local
        const float4 vv = *(const float4*)&Braw[buf][r * 64 + c4];
        BsT[c4 + 0][r] = f32_to_bf16(vv.x);
        BsT[c4 + 1][r] = f32_to_bf16(vv.y);
        BsT[c4 + 2][r] = f32_to_bf16(vv.z);
        BsT[c4 + 3][r] = f32_to_bf16(vv.w);
      }
    }
#else  // -------- fallback: direct global staging (fast + guarded paths) ----
    const bool kfull = (kt + 32) <= Kd;
    if (kt + 32 < Kd) {
      int pm = m0 + (tid >> 1);
      if (pm < M) __builtin_prefetch(A + (size_t)pm * lda + (kt + 32), 0, 1);
    }
    if (kfull && (m0 + 64) <= M) {
#pragma unroll
      for (int s = 0; s < 4; ++s) {
        int lin = s * 128 + tid;
        int r = lin >> 3, c4 = (lin & 7) << 2;
        const float4 vv = *(const float4*)(A + (size_t)(m0 + r) * lda + (kt + c4));
        *(unsigned int*)&As[r][c4]     = pack_bf16(vv.x, vv.y);
        *(unsigned int*)&As[r][c4 + 2] = pack_bf16(vv.z, vv.w);
      }
    } else {
#pragma unroll
      for (int s = 0; s < 16; ++s) {
        int lin = s * 128 + tid;
        int r = lin >> 5, c = lin & 31;
        int gm = m0 + r, gk = kt + c;
        float av = (gm < M && gk < Kd) ? A[(size_t)gm * lda + gk] : 0.f;
        As[r][c] = f32_to_bf16(av);
      }
    }
    if (kfull && (n0 + 64) <= Nn) {
      if (transB) {
#pragma unroll
        for (int s = 0; s < 4; ++s) {
          int lin = s * 128 + tid;
          int r = lin >> 3, c4 = (lin & 7) << 2;
          const float4 vv = *(const float4*)(B + (size_t)(n0 + r) * ldb + (kt + c4));
          *(unsigned int*)&BsT[r][c4]     = pack_bf16(vv.x, vv.y);
          *(unsigned int*)&BsT[r][c4 + 2] = pack_bf16(vv.z, vv.w);
        }
      } else {
#pragma unroll
        for (int s = 0; s < 4; ++s) {
          int lin = s * 128 + tid;
          int r = lin >> 4, c4 = (lin & 15) << 2;
          const float4 vv = *(const float4*)(B + (size_t)(kt + r) * ldb + (n0 + c4));
          BsT[c4 + 0][r] = f32_to_bf16(vv.x);
          BsT[c4 + 1][r] = f32_to_bf16(vv.y);
          BsT[c4 + 2][r] = f32_to_bf16(vv.z);
          BsT[c4 + 3][r] = f32_to_bf16(vv.w);
        }
      }
    } else {
#pragma unroll
      for (int s = 0; s < 16; ++s) {
        int lin = s * 128 + tid;
        int r = lin >> 5, c = lin & 31;   // r = n-local, c = k-local
        int gn = n0 + r, gk = kt + c;
        float bv;
        if (transB) bv = (gn < Nn && gk < Kd) ? B[(size_t)gn * ldb + gk] : 0.f;
        else        bv = (gk < Kd && gn < Nn) ? B[(size_t)gk * ldb + gn] : 0.f;
        BsT[r][c] = f32_to_bf16(bv);
      }
    }
#endif
    __syncthreads();

    // ---- fragments ----
    FragU fa[2], fb[2];
    const int khi = (lane >> 4) * 8;
#pragma unroll
    for (int p = 0; p < 8; ++p) {
      int k0 = ((p < 4) ? (2 * p) : (8 + 2 * p)) + khi;
      fa[0].u[p] = *(const unsigned int*)&As[wm + (lane & 15)][k0];
      fa[1].u[p] = *(const unsigned int*)&As[wm + 16 + (lane & 15)][k0];
    }
    const int kb = (lane >> 4) * 8;
#pragma unroll
    for (int p = 0; p < 8; ++p) {
      fb[0].u[p] = *(const unsigned int*)&BsT[wn + (lane & 15)][kb + ((p < 4) ? (2 * p) : (8 + 2 * p))];
      fb[1].u[p] = *(const unsigned int*)&BsT[wn + 16 + (lane & 15)][kb + ((p < 4) ? (2 * p) : (8 + 2 * p))];
    }

#pragma unroll
    for (int rt = 0; rt < 2; ++rt)
#pragma unroll
      for (int ct = 0; ct < 2; ++ct)
        acc[rt][ct] = __builtin_amdgcn_wmma_f32_16x16x32_bf16(
            false, fa[rt].v, false, fb[ct].v, (short)0, acc[rt][ct], false, false);

#if HAVE_TDM
    if (wv == 0) __builtin_amdgcn_s_wait_tensorcnt(0);  // next buffer landed
    __syncthreads();
    buf ^= 1;
#else
    __syncthreads();
#endif
  }

  // ---- store: C/D layout VGPR r -> M = r + 8*(lane>>4); N = lane&15 ----
#pragma unroll
  for (int rt = 0; rt < 2; ++rt) {
    int mb = m0 + wm + rt * 16 + ((lane >> 4) * 8);
#pragma unroll
    for (int ct = 0; ct < 2; ++ct) {
      int gn = n0 + wn + ct * 16 + (lane & 15);
      if (gn < Nn) {
#pragma unroll
        for (int r = 0; r < 8; ++r) {
          int gm = mb + r;
          if (gm < M) {
            size_t off = (size_t)gm * ldc + gn;
            if (accumulate) Cmat[off] += acc[rt][ct][r];
            else            Cmat[off]  = acc[rt][ct][r];
          }
        }
      }
    }
  }
}

// ---------------------------------------------------------------------------
// Row-wise L2 normalize (cols <= 256); one block per row
// ---------------------------------------------------------------------------
__global__ void rownorm_kernel(const float* __restrict__ in, float* __restrict__ out,
                               int rows, int cols) {
  __shared__ float red[256];
  int row = blockIdx.x, t = threadIdx.x;
  float x = 0.f;
  if (t < cols) x = in[(size_t)row * cols + t];
  red[t] = x * x;
  __syncthreads();
  for (int s = 128; s > 0; s >>= 1) { if (t < s) red[t] += red[t + s]; __syncthreads(); }
  float nrm = fmaxf(sqrtf(red[0]), 1e-12f);
  if (t < cols) out[(size_t)row * cols + t] = x / nrm;
}

// time gate + convex combine + l2norm; one block per row
__global__ void timegate_norm_kernel(const float* __restrict__ ent, const float* __restrict__ cur,
                                     const float* __restrict__ g, const float* __restrict__ tb,
                                     float* __restrict__ out, int rows, int cols) {
  __shared__ float red[256];
  int row = blockIdx.x, t = threadIdx.x;
  float v = 0.f;
  if (t < cols) {
    size_t o = (size_t)row * cols + t;
    float tw = 1.f / (1.f + __expf(-(g[o] + tb[t])));
    v = tw * cur[o] + (1.f - tw) * ent[o];
  }
  red[t] = v * v;
  __syncthreads();
  for (int s = 128; s > 0; s >>= 1) { if (t < s) red[t] += red[t + s]; __syncthreads(); }
  float nrm = fmaxf(sqrtf(red[0]), 1e-12f);
  if (t < cols) out[(size_t)row * cols + t] = v / nrm;
}

__global__ void zero_kernel(float* __restrict__ p, size_t n) {
  for (size_t i = (size_t)blockIdx.x * blockDim.x + threadIdx.x; i < n;
       i += (size_t)gridDim.x * blockDim.x) p[i] = 0.f;
}

// msg[e,d] = h[src[e],d] + rel[r[e],d]; hist rows are (src, r, dst) int32 triples
__global__ void gather_add_kernel(const float* __restrict__ h, const float* __restrict__ rel,
                                  const int* __restrict__ hist, float* __restrict__ out,
                                  int nE, int cols) {
  size_t total = (size_t)nE * cols;
  for (size_t idx = (size_t)blockIdx.x * blockDim.x + threadIdx.x; idx < total;
       idx += (size_t)gridDim.x * blockDim.x) {
    int e = (int)(idx / cols), d = (int)(idx % cols);
    int s = hist[(size_t)e * 3 + 0];
    int r = hist[(size_t)e * 3 + 1];
    out[idx] = h[(size_t)s * cols + d] + rel[(size_t)r * cols + d];
  }
}

__global__ void scatter_deg_kernel(const int* __restrict__ hist, float* __restrict__ deg, int nE) {
  int e = blockIdx.x * blockDim.x + threadIdx.x;
  if (e < nE) atomicAdd(&deg[hist[(size_t)e * 3 + 2]], 1.0f);
}

__global__ void scatter_add_kernel(const float* __restrict__ msg, const int* __restrict__ hist,
                                   float* __restrict__ agg, int nE, int cols) {
  size_t total = (size_t)nE * cols;
  for (size_t idx = (size_t)blockIdx.x * blockDim.x + threadIdx.x; idx < total;
       idx += (size_t)gridDim.x * blockDim.x) {
    int e = (int)(idx / cols), d = (int)(idx % cols);
    atomicAdd(&agg[(size_t)hist[(size_t)e * 3 + 2] * cols + d], msg[idx]);
  }
}

__global__ void rgcn_finalize_kernel(const float* __restrict__ agg, const float* __restrict__ deg,
                                     const float* __restrict__ selfm, float* __restrict__ out,
                                     int rows, int cols) {
  size_t total = (size_t)rows * cols;
  for (size_t idx = (size_t)blockIdx.x * blockDim.x + threadIdx.x; idx < total;
       idx += (size_t)gridDim.x * blockDim.x) {
    int r = (int)(idx / cols);
    out[idx] = agg[idx] / fmaxf(deg[r], 1.0f) + selfm[idx];
  }
}

// x0[q, 0, :] = ent[q_e]; x0[q, 1, :] = relation_raw[q_r]
__global__ void decoder_gather_kernel(const float* __restrict__ ent, const float* __restrict__ rel,
                                      const int* __restrict__ tgt, float* __restrict__ x0,
                                      int nq, int cols) {
  size_t total = (size_t)nq * cols;
  for (size_t idx = (size_t)blockIdx.x * blockDim.x + threadIdx.x; idx < total;
       idx += (size_t)gridDim.x * blockDim.x) {
    int q = (int)(idx / cols), d = (int)(idx % cols);
    int qe = tgt[(size_t)q * 3 + 0];
    int qr = tgt[(size_t)q * 3 + 1];
    x0[(size_t)q * 2 * cols + d]        = ent[(size_t)qe * cols + d];
    x0[(size_t)q * 2 * cols + cols + d] = rel[(size_t)qr * cols + d];
  }
}

// per-channel sum / sumsq for BN over layout (Q, cn, dn); channel = (idx/dn)%cn
__global__ void channel_stats_kernel(const float* __restrict__ x, float* __restrict__ sums,
                                     float* __restrict__ sumsq, long long total, int cn, int dn) {
  __shared__ float ss[256];
  __shared__ float sq[256];
  for (int i = threadIdx.x; i < cn; i += blockDim.x) { ss[i] = 0.f; sq[i] = 0.f; }
  __syncthreads();
  for (size_t idx = (size_t)blockIdx.x * blockDim.x + threadIdx.x; idx < (size_t)total;
       idx += (size_t)gridDim.x * blockDim.x) {
    int c = (int)((idx / dn) % cn);
    float v = x[idx];
    atomicAdd(&ss[c], v);
    atomicAdd(&sq[c], v * v);
  }
  __syncthreads();
  for (int i = threadIdx.x; i < cn; i += blockDim.x) {
    atomicAdd(&sums[i], ss[i]);
    atomicAdd(&sumsq[i], sq[i]);
  }
}

__global__ void bn_apply_kernel(float* __restrict__ x, const float* __restrict__ sums,
                                const float* __restrict__ sumsq, const float* __restrict__ gamma,
                                const float* __restrict__ beta, long long total, int cn, int dn,
                                float inv_count, int relu) {
  for (size_t idx = (size_t)blockIdx.x * blockDim.x + threadIdx.x; idx < (size_t)total;
       idx += (size_t)gridDim.x * blockDim.x) {
    int c = (int)((idx / dn) % cn);
    float m = sums[c] * inv_count;
    float var = sumsq[c] * inv_count - m * m;
    float y = (x[idx] - m) * rsqrtf(var + 1e-5f) * gamma[c] + beta[c];
    if (relu) y = fmaxf(y, 0.f);
    x[idx] = y;
  }
}

// 1x(2,K) conv over (Q,2,D) with width padding 1 each side; + bias
__global__ void conv_kernel(const float* __restrict__ x0, const float* __restrict__ w,
                            const float* __restrict__ bias, float* __restrict__ out,
                            int nq, int cc, int dd) {
  size_t total = (size_t)nq * cc * dd;
  for (size_t idx = (size_t)blockIdx.x * blockDim.x + threadIdx.x; idx < total;
       idx += (size_t)gridDim.x * blockDim.x) {
    int q = (int)(idx / ((size_t)cc * dd));
    int rem = (int)(idx % ((size_t)cc * dd));
    int c = rem / dd, d = rem % dd;
    const float* xq = x0 + (size_t)q * 2 * dd;
    float s = bias[c];
#pragma unroll
    for (int h = 0; h < 2; ++h) {
#pragma unroll
      for (int kk = 0; kk < KW; ++kk) {
        int di = d + kk - 1;
        if (di >= 0 && di < dd) s += xq[(size_t)h * dd + di] * w[(size_t)(c * 2 + h) * KW + kk];
      }
    }
    out[idx] = s;
  }
}

// ---------------------------------------------------------------------------
extern "C" void kernel_launch(void* const* d_in, const int* in_sizes, int n_in,
                              void* d_out, int out_size, void* d_ws, size_t ws_size,
                              hipStream_t stream) {
  (void)in_sizes; (void)n_in; (void)out_size;
  const float* entity   = (const float*)d_in[0];
  const float* relation = (const float*)d_in[1];
  const float* tg_w     = (const float*)d_in[2];
  const float* tg_b     = (const float*)d_in[3];
  const float* wn1      = (const float*)d_in[4];
  const float* ws1      = (const float*)d_in[5];
  const float* wn2      = (const float*)d_in[6];
  const float* ws2      = (const float*)d_in[7];
  const float* conv_w   = (const float*)d_in[8];
  const float* conv_b   = (const float*)d_in[9];
  const float* bn0_g    = (const float*)d_in[10];
  const float* bn0_b    = (const float*)d_in[11];
  const float* bn1_g    = (const float*)d_in[12];
  const float* bn1_b    = (const float*)d_in[13];
  const float* bn_g     = (const float*)d_in[14];
  const float* bn_b     = (const float*)d_in[15];
  const float* fc_w     = (const float*)d_in[16];
  const int*   hist     = (const int*)d_in[17];
  const int*   tgt      = (const int*)d_in[18];
  float*       score    = (float*)d_out;

  // Bump-allocate scratch from d_ws
  char* wp = (char*)d_ws;
  auto alloc = [&](size_t nf) { float* p = (float*)wp; wp += nf * sizeof(float); return p; };
  float* entA  = alloc((size_t)N_ENT * DIM);
  float* entB  = alloc((size_t)N_ENT * DIM);
  float* reln  = alloc((size_t)R2C * DIM);
  float* msgA  = alloc((size_t)NE * DIM);     // also reused as time-gate logits g
  float* msgB  = alloc((size_t)NE * DIM);     // also reused as cur = l2norm(h2)
  float* agg   = alloc((size_t)N_ENT * DIM);
  float* selfm = alloc((size_t)N_ENT * DIM);
  float* h1    = alloc((size_t)N_ENT * DIM);
  float* h2    = alloc((size_t)N_ENT * DIM);
  float* deg   = alloc((size_t)N_ENT);
  float* x0    = alloc((size_t)NQ * 2 * DIM);
  float* xc    = alloc((size_t)NQ * CCH * DIM);
  float* yb    = alloc((size_t)NQ * DIM);
  float* stat_s = alloc(256);
  float* stat_q = alloc(256);
  if ((size_t)(wp - (char*)d_ws) > ws_size) return;  // not enough scratch

  auto gsz = [](size_t n) {
    size_t g = (n + 255) / 256;
    return (unsigned)(g > 65535 ? 65535 : g);
  };
  auto zero = [&](float* p, size_t n) { zero_kernel<<<gsz(n), 256, 0, stream>>>(p, n); };
  auto gemm = [&](const float* A, const float* B, float* Cc, int M, int Nn, int Kd,
                  int lda, int ldb, int ldc, int tB, int accum) {
    dim3 grid((Nn + 63) / 64, (M + 63) / 64);
    gemm_bf16w<<<grid, 128, 0, stream>>>(A, B, Cc, M, Nn, Kd, lda, ldb, ldc, tB, accum);
  };
  auto rgcn = [&](const float* hin, const int* histj, const float* wn, const float* ws, float* hout) {
    gather_add_kernel<<<gsz((size_t)NE * DIM), 256, 0, stream>>>(hin, reln, histj, msgA, NE, DIM);
    gemm(msgA, wn, msgB, NE, DIM, DIM, DIM, DIM, DIM, 0, 0);
    zero(agg, (size_t)N_ENT * DIM);
    zero(deg, (size_t)N_ENT);
    scatter_deg_kernel<<<gsz(NE), 256, 0, stream>>>(histj, deg, NE);
    scatter_add_kernel<<<gsz((size_t)NE * DIM), 256, 0, stream>>>(msgB, histj, agg, NE, DIM);
    gemm(hin, ws, selfm, N_ENT, DIM, DIM, DIM, DIM, DIM, 0, 0);
    rgcn_finalize_kernel<<<gsz((size_t)N_ENT * DIM), 256, 0, stream>>>(agg, deg, selfm, hout, N_ENT, DIM);
  };

  zero(score, (size_t)NQ * N_ENT);

  for (int i = 0; i < LLY; ++i) {
    rownorm_kernel<<<N_ENT, 256, 0, stream>>>(entity, entA, N_ENT, DIM);
    rownorm_kernel<<<R2C, 256, 0, stream>>>(relation, reln, R2C, DIM);
    float* entCur = entA;
    float* entNxt = entB;
    for (int j = LLY - (i + 1); j < LLY; ++j) {
      const int* histj = hist + (size_t)j * NE * 3;
      rgcn(entCur, histj, wn1, ws1, h1);
      rgcn(h1, histj, wn2, ws2, h2);
      float* cur = msgB;  // reuse (free after scatter)
      float* g   = msgA;  // reuse (free after gemm)
      rownorm_kernel<<<N_ENT, 256, 0, stream>>>(h2, cur, N_ENT, DIM);
      gemm(entCur, tg_w, g, N_ENT, DIM, DIM, DIM, DIM, DIM, 0, 0);
      timegate_norm_kernel<<<N_ENT, 256, 0, stream>>>(entCur, cur, g, tg_b, entNxt, N_ENT, DIM);
      float* tmp = entCur; entCur = entNxt; entNxt = tmp;
    }

    // ---- decoder for step i ----
    decoder_gather_kernel<<<gsz((size_t)NQ * DIM), 256, 0, stream>>>(entCur, relation, tgt, x0, NQ, DIM);
    // bn0 over channel axis (2 channels), count = NQ*DIM
    zero(stat_s, 512);
    channel_stats_kernel<<<gsz((size_t)NQ * 2 * DIM), 256, 0, stream>>>(
        x0, stat_s, stat_q, (long long)NQ * 2 * DIM, 2, DIM);
    bn_apply_kernel<<<gsz((size_t)NQ * 2 * DIM), 256, 0, stream>>>(
        x0, stat_s, stat_q, bn0_g + (size_t)i * 2, bn0_b + (size_t)i * 2,
        (long long)NQ * 2 * DIM, 2, DIM, 1.f / ((float)NQ * DIM), 0);
    // conv + bias
    conv_kernel<<<gsz((size_t)NQ * CCH * DIM), 256, 0, stream>>>(
        x0, conv_w + (size_t)i * CCH * 2 * KW, conv_b + (size_t)i * CCH, xc, NQ, CCH, DIM);
    // bn1 (+relu), count = NQ*DIM per channel
    zero(stat_s, 512);
    channel_stats_kernel<<<gsz((size_t)NQ * CCH * DIM), 256, 0, stream>>>(
        xc, stat_s, stat_q, (long long)NQ * CCH * DIM, CCH, DIM);
    bn_apply_kernel<<<gsz((size_t)NQ * CCH * DIM), 256, 0, stream>>>(
        xc, stat_s, stat_q, bn1_g + (size_t)i * CCH, bn1_b + (size_t)i * CCH,
        (long long)NQ * CCH * DIM, CCH, DIM, 1.f / ((float)NQ * DIM), 1);
    // fc: (NQ, C*D) @ fc_w^T -> (NQ, D)
    gemm(xc, fc_w, yb, NQ, DIM, CCH * DIM, CCH * DIM, CCH * DIM, DIM, 1, 0);
    // final bn over feature axis (D channels), count = NQ, +relu
    zero(stat_s, 512);
    channel_stats_kernel<<<gsz((size_t)NQ * DIM), 256, 0, stream>>>(
        yb, stat_s, stat_q, (long long)NQ * DIM, DIM, 1);
    bn_apply_kernel<<<gsz((size_t)NQ * DIM), 256, 0, stream>>>(
        yb, stat_s, stat_q, bn_g + (size_t)i * DIM, bn_b + (size_t)i * DIM,
        (long long)NQ * DIM, DIM, 1, 1.f / (float)NQ, 1);
    // score += y @ entity_embed^T  (accumulating WMMA GEMM)
    gemm(yb, entity, score, NQ, N_ENT, DIM, DIM, DIM, N_ENT, 1, 1);
  }
}
